// FractalConsciousnessDynamics_55886114455823
// MI455X (gfx1250) — matrix operand order repaired and checked by reading, not compiled, not run on hardware
//
#include <hip/hip_runtime.h>
#include <hip/hip_bf16.h>

// ---------------------------------------------------------------------------
// FractalConsciousnessDynamics on MI455X (gfx1250, wave32, WMMA bf16)
//
// Pipeline (all on `stream`):
//   pack_weight x4 : fp32 W -> bf16 WMMA B-fragment layout in ws
//   gemm512<AF32,TANH> : bufA = tanh(x@W1+b1)            (bf16)
//   gemm512<BF16,LIN > : bufB = bufA@W2+b2               (bf16)
//   kuramoto10         : bufA = 10-step recurrence(bufB) (bf16)
//   gemm512<BF16,RELU> : bufB = relu(bufA@F1+c1)         (bf16)
//   gemm512<BF16,OUT4> : out[...,0] = bufB@F2+c2         (f32, stride 4)
//   imag_fill          : out[...,1..3] = noise*sin(alpha*e)
//
// GEMM inner loop: B fragments staged global->LDS with CDNA5 async-to-LDS
// (ASYNCcnt), double-buffered; A fragment double-buffered in registers.
// ---------------------------------------------------------------------------

typedef __attribute__((ext_vector_type(8)))  float  v8f;
typedef __attribute__((ext_vector_type(16))) __bf16 v16bf;

#define NTOK (8 * 4096)   // B*S tokens
#define DIM  512          // E == O == 512

// fp32 -> bf16, round-to-nearest-even
__device__ __forceinline__ unsigned int f2bf1(float f) {
  unsigned int u = __float_as_uint(f);
  return (u + 0x7FFFu + ((u >> 16) & 1u)) >> 16;
}
__device__ __forceinline__ unsigned int pack2bf(float lo, float hi) {
  return f2bf1(lo) | (f2bf1(hi) << 16);
}

union ABFrag { int4 q[2]; v16bf v; };

// CDNA5 async global->LDS copy (16 bytes), tracked by ASYNCcnt.
// lds_off: LDS byte offset (low 32 bits of generic pointer to __shared__).
__device__ __forceinline__ void async_b128(unsigned lds_off, unsigned long long gaddr) {
  asm volatile("global_load_async_to_lds_b128 %0, %1, off"
               :
               : "v"(lds_off), "v"(gaddr)
               : "memory");
}
__device__ __forceinline__ void wait_async0() {
  asm volatile("s_wait_asynccnt 0" ::: "memory");
}

// ---------------------------------------------------------------------------
// Repack one 512x512 fp32 row-major weight into bf16 B-fragment layout.
// Fragment (nt,kt) covers cols [16nt,16nt+16), rows [32kt,32kt+32).
// B 32x16 layout used here (contiguous k): lane = 16*k_half + n holds 16 consecutive K for
// its column n -> each lane's 16 bf16 stored contiguously (32 B / lane).
// fragIdx = nt*16 + kt  (kt contiguous -> streaming in GEMM K loop)
// ---------------------------------------------------------------------------
__global__ __launch_bounds__(256) void pack_weight(const float* __restrict__ W,
                                                   unsigned short* __restrict__ Wp) {
  const int lane = threadIdx.x & 31;
  const int frag = blockIdx.x * 8 + (threadIdx.x >> 5);  // 0..511
  const int nt   = frag >> 4;
  const int kt   = frag & 15;
  const int n    = nt * 16 + (lane & 15);
  const int k0   = kt * 32 + (lane >> 4) * 16;
  unsigned int d[8];
#pragma unroll
  for (int j = 0; j < 8; ++j) {
    float a = W[(size_t)(k0 + 2 * j)     * DIM + n];
    float b = W[(size_t)(k0 + 2 * j + 1) * DIM + n];
    d[j] = pack2bf(a, b);
  }
  int4* dst = (int4*)(Wp + (size_t)frag * 512);
  dst[lane * 2]     = make_int4((int)d[0], (int)d[1], (int)d[2], (int)d[3]);
  dst[lane * 2 + 1] = make_int4((int)d[4], (int)d[5], (int)d[6], (int)d[7]);
}

// ---------------------------------------------------------------------------
// A-fragment load (16-bit A 16x32, same contiguous k order as B).
// lanes 0-15: K [0..15]; lanes 16-31: K [16..31]
// ---------------------------------------------------------------------------
template <int AF32>
__device__ __forceinline__ ABFrag load_a(const void* __restrict__ Aptr, int mrow,
                                         int kg, int kt) {
  ABFrag a;
  if (AF32) {
    const float* ap = (const float*)Aptr + (size_t)mrow * DIM + kt * 32 + kg * 16;
    float4 f0 = ((const float4*)ap)[0];
    float4 f1 = ((const float4*)ap)[1];
    float4 f2 = ((const float4*)(ap + 8))[0];
    float4 f3 = ((const float4*)(ap + 8))[1];
    a.q[0] = make_int4((int)pack2bf(f0.x, f0.y), (int)pack2bf(f0.z, f0.w),
                       (int)pack2bf(f1.x, f1.y), (int)pack2bf(f1.z, f1.w));
    a.q[1] = make_int4((int)pack2bf(f2.x, f2.y), (int)pack2bf(f2.z, f2.w),
                       (int)pack2bf(f3.x, f3.y), (int)pack2bf(f3.z, f3.w));
  } else {
    const unsigned short* ap =
        (const unsigned short*)Aptr + (size_t)mrow * DIM + kt * 32 + kg * 16;
    a.q[0] = *(const int4*)ap;
    a.q[1] = *(const int4*)(ap + 8);
  }
  return a;
}

// ---------------------------------------------------------------------------
// Tiled bf16 WMMA GEMM: C[NTOK,512] = epilogue(A[NTOK,512] @ B[512,512] + bias)
// Block: 256 threads (8 waves). Block tile 128(M) x 128(N).
// Wave tile: 16(M) x 128(N) -> 8 C fragments, 128 WMMAs per wave.
// B fragments: async global->LDS (8 KB per K-step), double-buffered; every
// thread copies one 32 B chunk; waves read back with ds_load_b128.
// EPI: 0=tanh->bf16  1=linear->bf16  2=relu->bf16  3=f32 store stride 4
// ---------------------------------------------------------------------------
template <int AF32, int EPI>
__global__ __launch_bounds__(256) void gemm512(const void* __restrict__ Aptr,
                                               const unsigned short* __restrict__ Bp,
                                               const float* __restrict__ bias,
                                               void* __restrict__ Out) {
  __shared__ int4 smem[2][512];  // 2 x 8 KB B-fragment buffers

  const int lane = threadIdx.x & 31;
  const int wv   = threadIdx.x >> 5;          // 0..7
  const int rowb = blockIdx.x * 128 + wv * 16;
  const int colb = blockIdx.y * 128;
  const int mrow = rowb + (lane & 15);
  const int kg   = lane >> 4;                 // 0 or 1

  // staging-copy role: thread t moves 32 B of fragment nt_c at each K-step
  const int nt_c   = threadIdx.x >> 5;        // fragment 0..7
  const int lane_c = threadIdx.x & 31;
  const unsigned long long gbase =
      (unsigned long long)(const char*)Bp +
      ((unsigned long long)(colb / 16 + nt_c) * 16) * 1024ull + (unsigned)(lane_c * 32);
  const unsigned lds_c0 = (unsigned)(size_t)(const void*)&smem[0][nt_c * 64 + lane_c * 2];
  const unsigned lds_c1 = (unsigned)(size_t)(const void*)&smem[1][nt_c * 64 + lane_c * 2];

  v8f acc[8] = {};

  // prologue: stage kt=0 into buffer 0
  async_b128(lds_c0,      gbase);
  async_b128(lds_c0 + 16, gbase + 16);
  wait_async0();
  __syncthreads();

  ABFrag a = load_a<AF32>(Aptr, mrow, kg, 0);

  for (int kt = 0; kt < 16; ++kt) {
    const int cur = kt & 1;
    // issue async stage of kt+1 into the other buffer
    if (kt + 1 < 16) {
      const unsigned long long g = gbase + (unsigned)((kt + 1) * 1024);
      const unsigned l = (cur ? lds_c0 : lds_c1);
      async_b128(l,      g);
      async_b128(l + 16, g + 16);
    }
    // read this K-step's 8 B fragments from LDS
    ABFrag b[8];
#pragma unroll
    for (int nt = 0; nt < 8; ++nt) {
      b[nt].q[0] = smem[cur][nt * 64 + lane * 2];
      b[nt].q[1] = smem[cur][nt * 64 + lane * 2 + 1];
    }
    // double-buffer A in registers
    ABFrag a_n;
    if (kt + 1 < 16) a_n = load_a<AF32>(Aptr, mrow, kg, kt + 1);

#pragma unroll
    for (int nt = 0; nt < 8; ++nt) {
      acc[nt] = __builtin_amdgcn_wmma_f32_16x16x32_bf16(
          false, a.v, false, b[nt].v, (short)0, acc[nt], false, false);
    }

    if (kt + 1 < 16) {
      wait_async0();       // our async writes for kt+1 landed in LDS
      __syncthreads();     // everyone's writes visible; cur buffer free
      a = a_n;
    }
  }

  // ---- Epilogue: C layout VGPR v, lane l -> M = rowb + v + 8*(l>>4), N = l&15
#pragma unroll
  for (int nt = 0; nt < 8; ++nt) {
    const int col = colb + nt * 16 + (lane & 15);
    const float bv = bias[col];
#pragma unroll
    for (int v = 0; v < 8; ++v) {
      const int M = rowb + v + 8 * kg;
      float val = acc[nt][v] + bv;
      if (EPI == 0) val = tanhf(val);
      if (EPI == 2) val = fmaxf(val, 0.0f);
      if (EPI == 3) {
        ((float*)Out)[((size_t)M * DIM + col) * 4] = val;   // channel 0 of [.,4]
      } else {
        ((unsigned short*)Out)[(size_t)M * DIM + col] = (unsigned short)f2bf1(val);
      }
    }
  }
}

// ---------------------------------------------------------------------------
// 10-step Kuramoto recurrence. One wave per token: each lane keeps 16 phases
// in registers; full-row mean(sin) via lane-local sum + 5-step shfl_xor.
// ---------------------------------------------------------------------------
__global__ __launch_bounds__(256) void kuramoto10(const unsigned short* __restrict__ Ph,
                                                  unsigned short* __restrict__ Out,
                                                  const float* __restrict__ omega,
                                                  const float* __restrict__ Kp) {
  const int lane = threadIdx.x & 31;
  const int tok  = blockIdx.x * 8 + (threadIdx.x >> 5);

  const unsigned short* p = Ph + (size_t)tok * DIM + lane * 16;
  int4 r0 = ((const int4*)p)[0];
  int4 r1 = ((const int4*)p)[1];
  float ph[16];
  {
    unsigned int w[8] = {(unsigned)r0.x, (unsigned)r0.y, (unsigned)r0.z, (unsigned)r0.w,
                         (unsigned)r1.x, (unsigned)r1.y, (unsigned)r1.z, (unsigned)r1.w};
#pragma unroll
    for (int j = 0; j < 8; ++j) {
      ph[2 * j]     = __uint_as_float(w[j] << 16);
      ph[2 * j + 1] = __uint_as_float(w[j] & 0xFFFF0000u);
    }
  }
  float om[16];
  {
    const float4* op = (const float4*)(omega + lane * 16);
#pragma unroll
    for (int q = 0; q < 4; ++q) {
      float4 f = op[q];
      om[4 * q] = f.x; om[4 * q + 1] = f.y; om[4 * q + 2] = f.z; om[4 * q + 3] = f.w;
    }
  }
  const float Kv = Kp[0];
#pragma unroll
  for (int step = 0; step < 10; ++step) {
    float s = 0.0f;
#pragma unroll
    for (int j = 0; j < 16; ++j) s += __sinf(ph[j]);
#pragma unroll
    for (int off = 16; off > 0; off >>= 1) s += __shfl_xor(s, off, 32);
    const float mf = s * (Kv * (1.0f / 512.0f));
#pragma unroll
    for (int j = 0; j < 16; ++j) ph[j] += 0.01f * (om[j] + mf * __cosf(ph[j]));
  }
  unsigned int w[8];
#pragma unroll
  for (int j = 0; j < 8; ++j) w[j] = pack2bf(ph[2 * j], ph[2 * j + 1]);
  unsigned short* o = Out + (size_t)tok * DIM + lane * 16;
  ((int4*)o)[0] = make_int4((int)w[0], (int)w[1], (int)w[2], (int)w[3]);
  ((int4*)o)[1] = make_int4((int)w[4], (int)w[5], (int)w[6], (int)w[7]);
}

// ---------------------------------------------------------------------------
// Channels 1..3 of output: noise * sin(alpha * e)
// ---------------------------------------------------------------------------
__global__ __launch_bounds__(256) void imag_fill(const float* __restrict__ noise,
                                                 float* __restrict__ out,
                                                 const float* __restrict__ alphap) {
  const size_t i = (size_t)blockIdx.x * 256 + threadIdx.x;
  const int e = (int)(i & 511);
  const float v = noise[i] * __sinf(alphap[0] * (float)e);
  float* o = out + i * 4;
  o[1] = v;
  o[2] = v;
  o[3] = v;
}

// ---------------------------------------------------------------------------
extern "C" void kernel_launch(void* const* d_in, const int* in_sizes, int n_in,
                              void* d_out, int out_size, void* d_ws, size_t ws_size,
                              hipStream_t stream) {
  const float* x     = (const float*)d_in[0];
  const float* W1    = (const float*)d_in[1];
  const float* b1    = (const float*)d_in[2];
  const float* W2    = (const float*)d_in[3];
  const float* b2    = (const float*)d_in[4];
  const float* omega = (const float*)d_in[5];
  const float* Kp    = (const float*)d_in[6];
  const float* alpha = (const float*)d_in[7];
  const float* F1    = (const float*)d_in[8];
  const float* c1    = (const float*)d_in[9];
  const float* F2    = (const float*)d_in[10];
  const float* c2    = (const float*)d_in[11];
  const float* noise = (const float*)d_in[12];
  float* out = (float*)d_out;

  // ws layout: 4 packed weights (512 KB bf16 each) + two 32 MB bf16 buffers
  unsigned short* Wp   = (unsigned short*)d_ws;              // 4 * 262144 bf16
  unsigned short* bufA = Wp + 4 * 262144;                    // NTOK*512 bf16
  unsigned short* bufB = bufA + (size_t)NTOK * DIM;          // NTOK*512 bf16

  pack_weight<<<64, 256, 0, stream>>>(W1, Wp + 0 * 262144);
  pack_weight<<<64, 256, 0, stream>>>(W2, Wp + 1 * 262144);
  pack_weight<<<64, 256, 0, stream>>>(F1, Wp + 2 * 262144);
  pack_weight<<<64, 256, 0, stream>>>(F2, Wp + 3 * 262144);

  dim3 gg(NTOK / 128, DIM / 128);  // (256, 4)
  gemm512<1, 0><<<gg, 256, 0, stream>>>(x,    Wp + 0 * 262144, b1, bufA);  // tanh
  gemm512<0, 1><<<gg, 256, 0, stream>>>(bufA, Wp + 1 * 262144, b2, bufB);  // linear
  kuramoto10<<<NTOK / 8, 256, 0, stream>>>(bufB, bufA, omega, Kp);
  gemm512<0, 2><<<gg, 256, 0, stream>>>(bufA, Wp + 2 * 262144, c1, bufB);  // relu
  gemm512<0, 3><<<gg, 256, 0, stream>>>(bufB, Wp + 3 * 262144, c2, out);   // -> ch0
  imag_fill<<<(NTOK * DIM) / 256, 256, 0, stream>>>(noise, out, alpha);    // -> ch1..3
}
